// DICnet_19705309954348
// MI455X (gfx1250) — compile-verified, matrix operand order checked by reading, not run on hardware
//
#include <hip/hip_runtime.h>

// ---------------------------------------------------------------------------
// CDNA5 / gfx1250 implementation of the DIC flow network.
// Heavy convs = implicit GEMM on v_wmma_f32_16x16x32_f16 (wave32 WMMA),
// double-buffered LDS staging, 16x32 output tile per wave (2 WMMA / K-step),
// async global->LDS bias staging (ASYNCcnt) overlapped with the GEMM loop.
// ---------------------------------------------------------------------------

typedef __attribute__((ext_vector_type(16))) _Float16 v16h;
typedef __attribute__((ext_vector_type(8)))  float    v8f;

union AFrag { v16h v; _Float16 h[16]; unsigned u[8]; };
union CFrag { v8f  v; float f[8]; };

struct ConvP {
  const float* x; const float* w; const float* bias; float* y;
  int B, Cin, Hi, Wi, Cout, Ho, Wo;
  int k, stride, dil, pad;
  long xBS, yBS;   // per-batch element strides (supports channel-sliced concat buffers)
  int act;         // 0 = linear, 1 = leaky-relu(0.1)
};

#define BM 32
#define BN 128

#if defined(__gfx1250__) && \
    __has_builtin(__builtin_amdgcn_global_load_async_to_lds_b32) && \
    __has_builtin(__builtin_amdgcn_s_wait_asynccnt)
#define USE_ASYNC_BIAS 1
#endif

// Implicit-GEMM conv: M = Cout, N = B*Ho*Wo, K = Cin*k*k.
// Block = 256 threads = 8 waves in a 2(M) x 4(N) grid; each wave owns a
// 16(M) x 32(N) output tile -> two v_wmma per K-step sharing one A fragment.
__global__ __launch_bounds__(256) void conv_wmma_k(ConvP p) {
  __shared__ _Float16 sA[2][BM][40];        // 32(M) x 32(K) weights   (+pad)
  __shared__ _Float16 sB[2][32][BN + 8];    // 32(K) x 128(N) im2col   (+pad)
  __shared__ float    sBias[BM];

  const int tid  = threadIdx.x;
  const int lane = tid & 31;
  const int wid  = tid >> 5;
  const int wm   = wid >> 2;        // 0..1  (M tile)
  const int wn   = wid & 3;         // 0..3  (N tile, 32 cols each)
  const int HWo  = p.Ho * p.Wo;
  const int Npix = p.B * HWo;
  const int pixBase = blockIdx.x * BN;
  const int coBase  = blockIdx.y * BM;
  const int KK = p.k * p.k;
  const int K  = p.Cin * KK;

  // Bias tile: async DMA to LDS (tracked by ASYNCcnt), overlapped with the
  // whole GEMM loop; waited + barriered just before the epilogue.
#ifdef USE_ASYNC_BIAS
  if (wid == 0) {
    int co = coBase + lane;
    int cosafe = (co < p.Cout) ? co : (p.Cout - 1);   // keep address in-bounds
    __builtin_amdgcn_global_load_async_to_lds_b32(
        (__attribute__((address_space(1))) int*)(void*)(p.bias + cosafe),
        (__attribute__((address_space(3))) int*)&sBias[lane], 0, 0);
  }
#else
  if (tid < BM) {
    int co = coBase + tid;
    sBias[tid] = (co < p.Cout) ? p.bias[co] : 0.0f;
  }
#endif

  CFrag c0, c1;
  #pragma unroll
  for (int j = 0; j < 8; ++j) { c0.f[j] = 0.0f; c1.f[j] = 0.0f; }

  // Cooperative f32->f16 staging of one K-tile into LDS buffer `buf`.
  auto loadTile = [&](int buf, int kk) {
    // ---- A: weight tile 32(M) x 32(K), OIHW contiguous in K ----
    {
      int flat = tid * 4;              // 256 thr * 4 = 1024 = 32x32
      int row  = flat >> 5;
      int kc   = flat & 31;
      int co   = coBase + row;
      #pragma unroll
      for (int j = 0; j < 4; ++j) {
        int kidx = kk + kc + j;
        float v = 0.0f;
        if (co < p.Cout && kidx < K) v = p.w[(long)co * K + kidx];
        sA[buf][row][kc + j] = (_Float16)v;
      }
    }
    // ---- B: im2col tile 32(K) x 128(N), zero-pad at borders ----
    {
      int flat = tid * 16;             // 256 thr * 16 = 4096 = 32x128
      int krow = flat >> 7;
      int nc0  = flat & 127;
      int kidx = kk + krow;
      int ci = 0, r = 0, s = 0;
      bool kval = kidx < K;
      if (kval) { ci = kidx / KK; int rs = kidx - ci * KK; r = rs / p.k; s = rs - r * p.k; }
      long cbase = (long)ci * p.Hi * p.Wi;
      #pragma unroll
      for (int j = 0; j < 16; ++j) {
        int pix = pixBase + nc0 + j;
        float v = 0.0f;
        if (kval && pix < Npix) {
          int n  = pix / HWo; int hw = pix - n * HWo;
          int ho = hw / p.Wo, wo = hw - ho * p.Wo;
          int hi = ho * p.stride + r * p.dil - p.pad;
          int wi = wo * p.stride + s * p.dil - p.pad;
          if (hi >= 0 && hi < p.Hi && wi >= 0 && wi < p.Wi)
            v = p.x[(long)n * p.xBS + cbase + (long)hi * p.Wi + wi];
        }
        sB[buf][krow][nc0 + j] = (_Float16)v;
      }
    }
  };

  loadTile(0, 0);

  const int hs    = (lane >> 4) & 1;
  const int rA    = wm * 16 + (lane & 15);
  const int cB0   = wn * 32 + (lane & 15);   // first 16-col subtile
  const int cB1   = cB0 + 16;                // second 16-col subtile

  int cur = 0;
  for (int kk = 0; kk < K; kk += 32) {
    __syncthreads();                         // tile `cur` ready; prev reads done
    if (kk + 32 < K) {
      __builtin_prefetch(&p.w[(long)coBase * K + kk + 32], 0, 1);  // global_prefetch_b8
      loadTile(cur ^ 1, kk + 32);            // overlap gather with WMMA below
    }

    // ---- fragments per documented 16-bit WMMA VGPR layouts ----
    // A 16x32: lanes 0-15 M=lane, K={0..7,16..23}; lanes 16-31 M=lane-16, K={8..15,24..31}
    // B 32x16: mirrored (N across lanes, same K pattern per half)
    AFrag a, b0, b1;
    const unsigned* aRow = (const unsigned*)(&sA[cur][rA][0]);
    #pragma unroll
    for (int d = 0; d < 8; ++d) {
      int Kst = ((d < 4) ? (2 * d) : (2 * d + 8)) + 8 * hs;
      a.u[d]          = aRow[Kst >> 1];      // packed K,K+1
      b0.h[2 * d]     = sB[cur][Kst][cB0];
      b0.h[2 * d + 1] = sB[cur][Kst + 1][cB0];
      b1.h[2 * d]     = sB[cur][Kst][cB1];
      b1.h[2 * d + 1] = sB[cur][Kst + 1][cB1];
    }
    c0.v = __builtin_amdgcn_wmma_f32_16x16x32_f16(false, a.v, false, b0.v,
                                                  (short)0, c0.v, false, false);
    c1.v = __builtin_amdgcn_wmma_f32_16x16x32_f16(false, a.v, false, b1.v,
                                                  (short)0, c1.v, false, false);
    cur ^= 1;
  }

#ifdef USE_ASYNC_BIAS
  __builtin_amdgcn_s_wait_asynccnt(0);       // wave 0 drains its async copy
#endif
  __syncthreads();                           // sBias visible to all waves

  // ---- epilogue: C VGPR j holds M=j (lanes 0-15) / M=j+8 (lanes 16-31) ----
  const int nl = lane & 15;
  #pragma unroll
  for (int half = 0; half < 2; ++half) {
    const int pix = pixBase + wn * 32 + half * 16 + nl;
    CFrag& c = half ? c1 : c0;
    if (pix < Npix) {
      const int n  = pix / HWo; const int hw = pix - n * HWo;
      #pragma unroll
      for (int j = 0; j < 8; ++j) {
        int co = coBase + wm * 16 + j + 8 * hs;
        if (co < p.Cout) {
          float v = c.f[j] + sBias[co - coBase];
          if (p.act) v = (v >= 0.0f) ? v : 0.1f * v;
          p.y[(long)n * p.yBS + (long)co * HWo + hw] = v;
        }
      }
    }
  }
}

// ---------------- BatchNorm (training-mode batch stats) --------------------
__global__ void bn_stats_k(const float* x, float* stats, int B, int C, int HW) {
  int c = blockIdx.x;
  long n = (long)B * HW;
  float s = 0.f, s2 = 0.f;
  for (long i = threadIdx.x; i < n; i += blockDim.x) {
    long b = i / HW, hw = i - b * HW;
    float v = x[(b * C + c) * (long)HW + hw];
    s += v; s2 += v * v;
  }
  __shared__ float sh[256], sh2[256];
  sh[threadIdx.x] = s; sh2[threadIdx.x] = s2;
  __syncthreads();
  for (int o = 128; o > 0; o >>= 1) {
    if ((int)threadIdx.x < o) { sh[threadIdx.x] += sh[threadIdx.x + o]; sh2[threadIdx.x] += sh2[threadIdx.x + o]; }
    __syncthreads();
  }
  if (threadIdx.x == 0) {
    float m   = sh[0] / (float)n;
    float var = sh2[0] / (float)n - m * m;
    stats[c]     = m;
    stats[C + c] = rsqrtf(var + 1e-5f);
  }
}

__global__ void bn_apply_k(const float* x, float* y, const float* stats,
                           const float* g, const float* bb, int C, int HW,
                           long total, int act) {
  long i = blockIdx.x * (long)blockDim.x + threadIdx.x;
  if (i >= total) return;
  int c = (int)((i / HW) % C);
  float v = (x[i] - stats[c]) * stats[C + c] * g[c] + bb[c];
  if (act) v = (v >= 0.f) ? v : 0.1f * v;
  y[i] = v;
}

// ---------------- elementwise helpers --------------------------------------
__global__ void fill0_k(float* y, long n) {
  long i = blockIdx.x * (long)blockDim.x + threadIdx.x;
  if (i < n) y[i] = 0.f;
}
__global__ void addlrelu_k(const float* a, const float* b, float* y, long n) {
  long i = blockIdx.x * (long)blockDim.x + threadIdx.x;
  if (i >= n) return;
  float v = a[i] + b[i];
  y[i] = (v >= 0.f) ? v : 0.1f * v;
}
__global__ void add3_k(const float* a, const float* b, const float* c, float* y, long n) {
  long i = blockIdx.x * (long)blockDim.x + threadIdx.x;
  if (i < n) y[i] = a[i] + b[i] + c[i];
}
__global__ void add2s_k(const float* a, long aBS, const float* b2, long bBS,
                        float* y, long yBS, long perB, long total) {
  long i = blockIdx.x * (long)blockDim.x + threadIdx.x;
  if (i >= total) return;
  long bi = i / perB, r = i - bi * perB;
  y[bi * yBS + r] = a[bi * aBS + r] + b2[bi * bBS + r];
}
__global__ void copych_k(const float* x, long xBS, float* y, long yBS,
                         long perB, long total) {
  long i = blockIdx.x * (long)blockDim.x + threadIdx.x;
  if (i >= total) return;
  long bi = i / perB, r = i - bi * perB;
  y[bi * yBS + r] = x[bi * xBS + r];
}

// ---------------- correlation (cost volume), fused /C and lrelu ------------
__global__ void corr_k(const float* f1, const float* f2, float* y,
                       int B, int C, int H, int W, long yBS) {
  int HW = H * W;
  int hw = blockIdx.x * blockDim.x + threadIdx.x;
  if (hw >= HW) return;
  int d = blockIdx.y, b = blockIdx.z;
  int dy = d / 9 - 4, dx = d - (d / 9) * 9 - 4;
  int h = hw / W, w = hw - h * W;
  int h2 = h + dy, w2 = w + dx;
  float s = 0.f;
  if (h2 >= 0 && h2 < H && w2 >= 0 && w2 < W) {
    const float* p1 = f1 + (long)b * C * HW + hw;
    const float* p2 = f2 + (long)b * C * HW + (long)h2 * W + w2;
    for (int c = 0; c < C; ++c) s += p1[(long)c * HW] * p2[(long)c * HW];
  }
  s /= (float)C;
  s = (s >= 0.f) ? s : 0.1f * s;
  y[(long)b * yBS + (long)d * HW + hw] = s;
}

// ---------------- bilinear warp with zeros + >=1 mask ----------------------
__global__ void warp_k(const float* x, const float* flow, float* y,
                       int B, int C, int H, int W) {
  long total = (long)B * C * H * W;
  long i = blockIdx.x * (long)blockDim.x + threadIdx.x;
  if (i >= total) return;
  int w = (int)(i % W); long t = i / W;
  int h = (int)(t % H); t /= H;
  int c = (int)(t % C); int b = (int)(t / C);
  long HW = (long)H * W; long hw = (long)h * W + w;
  float fx = flow[((long)b * 2 + 0) * HW + hw];
  float fy = flow[((long)b * 2 + 1) * HW + hw];
  float px = (float)w + fx, py = (float)h + fy;
  float x0f = floorf(px), y0f = floorf(py);
  float wx1 = px - x0f, wy1 = py - y0f;
  int x0 = (int)x0f, y0 = (int)y0f;
  float acc = 0.f, wsum = 0.f;
  #pragma unroll
  for (int dy = 0; dy < 2; ++dy)
    #pragma unroll
    for (int dx = 0; dx < 2; ++dx) {
      int yi = y0 + dy, xi = x0 + dx;
      float wgt = (dy ? wy1 : 1.f - wy1) * (dx ? wx1 : 1.f - wx1);
      bool inb = (yi >= 0 && yi < H && xi >= 0 && xi < W);
      float wv = inb ? wgt : 0.f;
      int yc = yi < 0 ? 0 : (yi > H - 1 ? H - 1 : yi);
      int xc = xi < 0 ? 0 : (xi > W - 1 ? W - 1 : xi);
      acc  += x[((long)(b * C + c) * H + yc) * W + xc] * wv;
      wsum += wv;
    }
  y[i] = (wsum >= 1.0f) ? acc : 0.0f;
}

// ---------------- 2x bilinear flow upsample (half-pixel) * 2.0 -------------
__global__ void upflow_k(const float* fin, float* fout, int B,
                         int Hi, int Wi, int Ho, int Wo) {
  long total = (long)B * 2 * Ho * Wo;
  long i = blockIdx.x * (long)blockDim.x + threadIdx.x;
  if (i >= total) return;
  int wo = (int)(i % Wo); long t = i / Wo;
  int ho = (int)(t % Ho); t /= Ho;
  int ch = (int)(t % 2); int b = (int)(t / 2);
  float sy = (float)Hi / (float)Ho, sx = (float)Wi / (float)Wo;
  float py = (ho + 0.5f) * sy - 0.5f;
  float px = (wo + 0.5f) * sx - 0.5f;
  int y0 = (int)floorf(py), x0 = (int)floorf(px);
  float wy = py - (float)y0, wx = px - (float)x0;
  float acc = 0.f;
  #pragma unroll
  for (int dy = 0; dy < 2; ++dy)
    #pragma unroll
    for (int dx = 0; dx < 2; ++dx) {
      int yi = y0 + dy; yi = yi < 0 ? 0 : (yi > Hi - 1 ? Hi - 1 : yi);
      int xi = x0 + dx; xi = xi < 0 ? 0 : (xi > Wi - 1 ? Wi - 1 : xi);
      float wgt = (dy ? wy : 1.f - wy) * (dx ? wx : 1.f - wx);
      acc += fin[((long)(b * 2 + ch) * Hi + yi) * Wi + xi] * wgt;
    }
  fout[i] = 2.0f * acc;
}

// ===========================================================================
// Host orchestration
// ===========================================================================
struct CPW { const float* b; const float* w; };
struct BNW { const float* b; const float* g; };
struct DenseW { CPW c[5]; CPW convs[7]; CPW last; };
struct RBW { CPW c1, c2, down; BNW n1, n2, n3; };

extern "C" void kernel_launch(void* const* d_in, const int* in_sizes, int n_in,
                              void* d_out, int out_size, void* d_ws, size_t ws_size,
                              hipStream_t stream) {
  (void)in_sizes; (void)n_in; (void)out_size;
  const int B = 2;
  const int Hl[6] = {256, 128, 64, 32, 16, 8};
  const int Cl[6] = {16, 32, 64, 96, 128, 196};

  // ---- parameter walk: sorted pytree leaf order ('dense' < 'enc'),
  //      dict keys sorted ('b' < 'g', 'b' < 'w', c1..c5 < convs < last,
  //      c1 < c2 < down < n1 < n2 < n3) ----
  int cur = 2;
  auto NP = [&]() { return (const float*)d_in[cur++]; };
  DenseW dw[5];
  for (int i = 0; i < 5; ++i) {
    for (int j = 0; j < 5; ++j) { dw[i].c[j].b = NP(); dw[i].c[j].w = NP(); }
    for (int j = 0; j < 7; ++j) { dw[i].convs[j].b = NP(); dw[i].convs[j].w = NP(); }
    dw[i].last.b = NP(); dw[i].last.w = NP();
  }
  CPW conv1; conv1.b = NP(); conv1.w = NP();
  RBW rb[6][2];
  for (int g = 0; g < 6; ++g)
    for (int r = 0; r < 2; ++r) {
      bool hd = (r == 0 && g > 0);   // first resblock of groups 1..5 downsamples
      RBW& q = rb[g][r];
      q.c1.b = NP(); q.c1.w = NP();
      q.c2.b = NP(); q.c2.w = NP();
      if (hd) { q.down.b = NP(); q.down.w = NP(); }
      q.n1.b = NP(); q.n1.g = NP();
      q.n2.b = NP(); q.n2.g = NP();
      if (hd) { q.n3.b = NP(); q.n3.g = NP(); }
    }
  BNW encn1; encn1.b = NP(); encn1.g = NP();

  // ---- workspace carve ----
  char* base = (char*)d_ws;
  size_t off = 0;
  auto AL = [&](long floats) -> float* {
    float* p = (float*)(base + off);
    off += (size_t)((floats * 4 + 255) / 256) * 256;
    return p;
  };
  float* feats[2][6];
  for (int im = 0; im < 2; ++im)
    for (int l = 0; l < 6; ++l)
      feats[im][l] = AL((long)B * Cl[l] * Hl[l] * Hl[l]);
  float* scr[6];
  for (int i = 0; i < 6; ++i) scr[i] = AL((long)B * 16 * 256 * 256);
  float* stats = AL(512);
  float* big   = AL((long)B * 565 * 128 * 128);   // max Ctot (=115+450) @ 128x128
  float* ping  = AL((long)B * 128 * 128 * 128);
  float* pong  = AL((long)B * 128 * 128 * 128);
  float* xout  = AL((long)B * 2 * 128 * 128);
  float* resv  = AL((long)B * 2 * 128 * 128);
  float* fUp   = AL((long)B * 2 * 128 * 128);
  float* fA    = AL((long)B * 2 * 128 * 128);
  float* fBv   = AL((long)B * 2 * 128 * 128);
  float* bw    = AL((long)B * 32 * 128 * 128);    // max warped feature (li=1)
  if (off > ws_size) {  // workspace too small: emit deterministic zeros
    long n = (long)B * 2 * 128 * 128;
    fill0_k<<<(unsigned)((n + 255) / 256), 256, 0, stream>>>((float*)d_out, n);
    return;
  }

  auto CONV = [&](const float* x, const CPW& pw, float* y,
                  int Cin, int Hi, int Cout, int k, int stride, int dil,
                  long xBS, long yBS, int act) {
    int pad = (k - 1) * dil / 2;
    int span = (k - 1) * dil + 1;
    int Ho = (Hi + 2 * pad - span) / stride + 1;
    ConvP p;
    p.x = x; p.w = pw.w; p.bias = pw.b; p.y = y;
    p.B = B; p.Cin = Cin; p.Hi = Hi; p.Wi = Hi; p.Cout = Cout; p.Ho = Ho; p.Wo = Ho;
    p.k = k; p.stride = stride; p.dil = dil; p.pad = pad;
    p.xBS = xBS; p.yBS = yBS; p.act = act;
    dim3 gr((unsigned)((B * Ho * Ho + BN - 1) / BN), (unsigned)((Cout + BM - 1) / BM));
    conv_wmma_k<<<gr, 256, 0, stream>>>(p);
  };
  auto BN_ = [&](const float* x, float* y, const BNW& nb, int C, int HW, int act) {
    bn_stats_k<<<dim3(C), dim3(256), 0, stream>>>(x, stats, B, C, HW);
    long total = (long)B * C * HW;
    bn_apply_k<<<(unsigned)((total + 255) / 256), 256, 0, stream>>>(
        x, y, stats, nb.g, nb.b, C, HW, total, act);
  };

  // ---- encoder (both images) ----
  for (int im = 0; im < 2; ++im) {
    const float* img = (const float*)d_in[im];
    CONV(img, conv1, scr[0], 1, 512, 16, 7, 2, 1, (long)512 * 512, (long)16 * 256 * 256, 0);
    BN_(scr[0], scr[1], encn1, 16, 256 * 256, 1);
    const float* x = scr[1];
    int Cc = 16, Hc = 256;
    for (int g = 0; g < 6; ++g) {
      int Cout = Cl[g], s = (g == 0) ? 1 : 2;
      int Hn = Hc / s;
      long inBS = (long)Cc * Hc * Hc, outBS = (long)Cout * Hn * Hn;
      const RBW& r0 = rb[g][0];
      CONV(x, r0.c1, scr[2], Cc, Hc, Cout, 3, s, 1, inBS, outBS, 0);
      BN_(scr[2], scr[3], r0.n1, Cout, Hn * Hn, 1);
      CONV(scr[3], r0.c2, scr[2], Cout, Hn, Cout, 3, 1, 1, outBS, outBS, 0);
      BN_(scr[2], scr[4], r0.n2, Cout, Hn * Hn, 1);
      const float* xd = x;
      if (g > 0) {
        CONV(x, r0.down, scr[2], Cc, Hc, Cout, 1, s, 1, inBS, outBS, 0);
        BN_(scr[2], scr[5], r0.n3, Cout, Hn * Hn, 0);
        xd = scr[5];
      }
      long n = (long)B * Cout * Hn * Hn;
      addlrelu_k<<<(unsigned)((n + 255) / 256), 256, 0, stream>>>(xd, scr[4], scr[0], n);
      const RBW& r1 = rb[g][1];
      CONV(scr[0], r1.c1, scr[2], Cout, Hn, Cout, 3, 1, 1, outBS, outBS, 0);
      BN_(scr[2], scr[3], r1.n1, Cout, Hn * Hn, 1);
      CONV(scr[3], r1.c2, scr[2], Cout, Hn, Cout, 3, 1, 1, outBS, outBS, 0);
      BN_(scr[2], scr[4], r1.n2, Cout, Hn * Hn, 1);
      addlrelu_k<<<(unsigned)((n + 255) / 256), 256, 0, stream>>>(scr[0], scr[4], feats[im][g], n);
      x = feats[im][g]; Cc = Cout; Hc = Hn;
    }
  }

  // ---- coarse-to-fine pyramid (levels 6..2 => feats[5]..feats[1]) ----
  float* flowPrev = fA;
  float* flowCur  = fBv;
  for (int idx = 0; idx < 5; ++idx) {
    int li = 5 - idx;
    int C = Cl[li], H = Hl[li], HW = H * H;
    const float* a  = feats[0][li];
    const float* bf = feats[1][li];
    const float* bwp;
    if (idx == 0) {
      long n = (long)B * 2 * HW;
      fill0_k<<<(unsigned)((n + 255) / 256), 256, 0, stream>>>(fUp, n);
      bwp = bf;
    } else {
      int Hp = Hl[li + 1];
      long n = (long)B * 2 * HW;
      upflow_k<<<(unsigned)((n + 255) / 256), 256, 0, stream>>>(flowPrev, fUp, B, Hp, Hp, H, H);
      long nw = (long)B * C * HW;
      warp_k<<<(unsigned)((nw + 255) / 256), 256, 0, stream>>>(bf, fUp, bw, B, C, H, H);
      bwp = bw;
    }
    int ch_in = 83 + C;                  // 81 corr + C feat + 2 flow
    int Ctot  = ch_in + 450;             // + dense growth (448) + tail (2)
    long bigBS = (long)Ctot * HW;
    // feat = concat([corr, a, flow_up]) built in-place at channel offset 448
    corr_k<<<dim3((unsigned)((HW + 255) / 256), 81, B), 256, 0, stream>>>(
        a, bwp, big + (long)448 * HW, B, C, H, H, bigBS);
    { long n = (long)B * C * HW;
      copych_k<<<(unsigned)((n + 255) / 256), 256, 0, stream>>>(
          a, (long)C * HW, big + (long)(448 + 81) * HW, bigBS, (long)C * HW, n); }
    { long n = (long)B * 2 * HW;
      copych_k<<<(unsigned)((n + 255) / 256), 256, 0, stream>>>(
          fUp, (long)2 * HW, big + (long)(448 + 81 + C) * HW, bigBS, (long)2 * HW, n); }

    const DenseW& dp = dw[idx];
    int offc = 448, N = ch_in;
    const int fcs[5] = {128, 128, 96, 64, 32};
    for (int j = 0; j < 5; ++j) {        // x = concat([lrelu(conv(x)), x]) via prepend
      int fc = fcs[j];
      CONV(big + (long)offc * HW, dp.c[j], big + (long)(offc - fc) * HW,
           N, H, fc, 3, 1, 1, bigBS, bigBS, 1);
      offc -= fc; N += fc;
    }
    CONV(big, dp.last, xout, N, H, 2, 3, 1, 1, bigBS, (long)2 * HW, 0);  // x_out
    { long n = (long)B * 2 * HW;         // h tail = disp + x_out
      add2s_k<<<(unsigned)((n + 255) / 256), 256, 0, stream>>>(
          fUp, (long)2 * HW, xout, (long)2 * HW,
          big + (long)N * HW, bigBS, (long)2 * HW, n); }
    const int douts[7] = {128, 128, 128, 96, 64, 32, 2};
    const int dils[7]  = {1, 2, 4, 8, 16, 1, 1};
    const float* cp_ = big; long cBS = bigBS; int cinC = N + 2;
    for (int j = 0; j < 7; ++j) {
      float* outp = (j == 6) ? resv : ((j & 1) ? pong : ping);
      CONV(cp_, dp.convs[j], outp, cinC, H, douts[j], 3, 1, dils[j],
           cBS, (long)douts[j] * HW, (j < 6) ? 1 : 0);
      cp_ = outp; cBS = (long)douts[j] * HW; cinC = douts[j];
    }
    { long n = (long)B * 2 * HW;         // flow = disp + (h + x_out)
      add3_k<<<(unsigned)((n + 255) / 256), 256, 0, stream>>>(fUp, resv, xout, flowCur, n); }
    float* t = flowPrev; flowPrev = flowCur; flowCur = t;
  }

  (void)hipMemcpyAsync(d_out, flowPrev, (size_t)B * 2 * 128 * 128 * sizeof(float),
                       hipMemcpyDeviceToDevice, stream);
}
